// Decoder_14516989461057
// MI455X (gfx1250) — matrix operand order repaired and checked by reading, not compiled, not run on hardware
//
#include <hip/hip_runtime.h>

typedef __attribute__((ext_vector_type(16))) __bf16 v16bf;
typedef __attribute__((ext_vector_type(8)))  __bf16 v8bf;
typedef __attribute__((ext_vector_type(8)))  float  v8f;

namespace {
constexpr int kB   = 512;
constexpr int kH   = 1024;
constexpr int kNC  = 128;
constexpr int kT   = 256;
constexpr int k4H  = 4096;
constexpr int kInW = kNC + kH;   // 1152 columns of W_ih
}

__device__ __forceinline__ float sigf(float x) {
  return 1.0f / (1.0f + __expf(-x));
}
__device__ __forceinline__ float tanhfast(float x) {
  x = fminf(fmaxf(x, -15.0f), 15.0f);
  float e = __expf(2.0f * x);
  return (e - 1.0f) / (e + 1.0f);
}

// A-matrix 16x32 bf16 fragment (M rows striped across lanes 0-15 == 16-31):
// lanes 0-15 hold K = [0..15]; lanes 16-31 hold K = [16..31].
__device__ __forceinline__ v16bf load_a_frag(const __bf16* base, int lh) {
  v8bf lo = *(const v8bf*)(base + lh * 16);
  v8bf hi = *(const v8bf*)(base + lh * 16 + 8);
  return __builtin_shufflevector(lo, hi, 0,1,2,3,4,5,6,7,8,9,10,11,12,13,14,15);
}
// B-matrix 32x16 bf16 fragment: lane n holds K=[0..15] (lanes 0-15) or
// K=[16..31] (lanes 16-31) of column n. base points at row-major W row (= output col).
__device__ __forceinline__ v16bf load_b_frag(const __bf16* base, int lh) {
  return *(const v16bf*)(base + lh * 16);
}

// ---------------- one-time prep kernels ----------------

__global__ void prep_combine_w(const float* __restrict__ W_ih,
                               const float* __restrict__ W_hh,
                               __bf16* __restrict__ Wc) {
  int i = blockIdx.x * blockDim.x + threadIdx.x;     // over 4H*H
  if (i >= k4H * kH) return;
  int n = i / kH, k = i - n * kH;
  float v = W_ih[(size_t)n * kInW + kNC + k] + W_hh[i];
  Wc[i] = (__bf16)v;
}

__global__ void prep_wout(const float* __restrict__ W_out, __bf16* __restrict__ Wo) {
  int i = blockIdx.x * blockDim.x + threadIdx.x;     // over NC*H
  if (i >= kNC * kH) return;
  Wo[i] = (__bf16)W_out[i];
}

__global__ void prep_xpart(const float* __restrict__ inp0,
                           const float* __restrict__ W_ih,
                           const float* __restrict__ b_ih,
                           const float* __restrict__ b_hh,
                           float* __restrict__ xp) {
  int i = blockIdx.x * blockDim.x + threadIdx.x;     // over B*4H
  if (i >= kB * k4H) return;
  int b = i / k4H, n = i - b * k4H;
  float acc = b_ih[n] + b_hh[n];
  const float* wrow = W_ih + (size_t)n * kInW;
  const float* x = inp0 + (size_t)b * kNC;
  #pragma unroll 8
  for (int j = 0; j < kNC; ++j) acc += x[j] * wrow[j];
  xp[i] = acc;
}

__global__ void prep_state(const float* __restrict__ hid,
                           const float* __restrict__ cell0,
                           __bf16* __restrict__ hbf, float* __restrict__ c) {
  int i = blockIdx.x * blockDim.x + threadIdx.x;     // over B*H
  if (i >= kB * kH) return;
  hbf[i] = (__bf16)hid[i];
  c[i] = cell0[i];
}

// ---------------- per-step fused gate GEMM + LSTM cell ----------------
// Wave tile: M=32 batch rows x 16 hidden columns, computed simultaneously for
// all 4 gate regions (i,f,g,o) so the cell update fuses in registers.
// Block = 8 waves arranged 4 (m) x 2 (hc): co-resident waves share both the
// 128-row h slab and the 4 KB/k-step W column tiles through the WGP cache,
// cutting per-step L2 traffic to ~64 MB.
// grid = 128 blocks x 256 threads -> 1024 waves = 16 m-tiles * 64 col-tiles.

__global__ __launch_bounds__(256) void lstm_gate_step(
    const __bf16* __restrict__ h_in, const __bf16* __restrict__ Wc,
    const float* __restrict__ xp, float* __restrict__ c,
    __bf16* __restrict__ h_out) {
  const int wib   = threadIdx.x >> 5;          // 0..7
  const int lane  = threadIdx.x & 31;
  const int lr    = lane & 15;
  const int lh    = lane >> 4;
  const int mtile = (blockIdx.x & 3) * 4 + (wib & 3);   // 0..15
  const int hc    = (blockIdx.x >> 2) * 2 + (wib >> 2); // 0..63
  const int m0    = mtile * 32;
  const int col0  = hc * 16;

  // Accumulators initialized with x_part (acts as WMMA C operand).
  v8f acc[2][4];
  #pragma unroll
  for (int mi = 0; mi < 2; ++mi) {
    const float* p = xp + (size_t)(m0 + mi * 16 + lh * 8) * k4H + col0 + lr;
    #pragma unroll
    for (int g = 0; g < 4; ++g)
      #pragma unroll
      for (int r = 0; r < 8; ++r)
        acc[mi][g][r] = p[(size_t)r * k4H + g * kH];
  }

  for (int k0 = 0; k0 < kH; k0 += 32) {
    v16bf a[2], b[4];
    #pragma unroll
    for (int mi = 0; mi < 2; ++mi)
      a[mi] = load_a_frag(h_in + (size_t)(m0 + mi * 16 + lr) * kH + k0, lh);
    #pragma unroll
    for (int g = 0; g < 4; ++g)
      b[g] = load_b_frag(Wc + (size_t)(g * kH + col0 + lr) * kH + k0, lh);
    #pragma unroll
    for (int mi = 0; mi < 2; ++mi)
      #pragma unroll
      for (int g = 0; g < 4; ++g)
        acc[mi][g] = __builtin_amdgcn_wmma_f32_16x16x32_bf16(
            false, a[mi], false, b[g], (short)0, acc[mi][g], false, false);
  }

  // Fused LSTM cell update: i,f,g,o fragments align on (row, hidden-col).
  #pragma unroll
  for (int mi = 0; mi < 2; ++mi) {
    #pragma unroll
    for (int r = 0; r < 8; ++r) {
      int row = m0 + mi * 16 + lh * 8 + r;
      size_t ci = (size_t)row * kH + col0 + lr;
      float iv = acc[mi][0][r];
      float fv = acc[mi][1][r];
      float gv = acc[mi][2][r];
      float ov = acc[mi][3][r];
      float cn = sigf(fv) * c[ci] + sigf(iv) * tanhfast(gv);
      float hn = sigf(ov) * tanhfast(cn);
      c[ci] = cn;                 // c owned by exactly this wave -> in-place OK
      h_out[ci] = (__bf16)hn;     // h double-buffered (GEMM reads h_in)
    }
  }
}

// ---------------- per-step logits GEMM: out[:, :, t] = h @ W_out^T + b_out ----
// Wave tile 32x32; 64 waves = 8 blocks x 8 waves. 16 m-tiles * 4 n-tiles.

__global__ __launch_bounds__(256) void lstm_logits_step(
    const __bf16* __restrict__ h, const __bf16* __restrict__ Wo,
    const float* __restrict__ b_out, float* __restrict__ out, int t) {
  const int wave  = blockIdx.x * (blockDim.x >> 5) + (threadIdx.x >> 5);
  const int lane  = threadIdx.x & 31;
  const int lr    = lane & 15;
  const int lh    = lane >> 4;
  const int mtile = wave >> 2;          // 0..15
  const int ntile = wave & 3;           // 0..3
  const int m0    = mtile * 32;
  const int n0    = ntile * 32;

  v8f acc[2][2] = {};
  for (int k0 = 0; k0 < kH; k0 += 32) {
    v16bf a[2], b[2];
    #pragma unroll
    for (int mi = 0; mi < 2; ++mi)
      a[mi] = load_a_frag(h + (size_t)(m0 + mi * 16 + lr) * kH + k0, lh);
    #pragma unroll
    for (int ni = 0; ni < 2; ++ni)
      b[ni] = load_b_frag(Wo + (size_t)(n0 + ni * 16 + lr) * kH + k0, lh);
    #pragma unroll
    for (int mi = 0; mi < 2; ++mi)
      #pragma unroll
      for (int ni = 0; ni < 2; ++ni)
        acc[mi][ni] = __builtin_amdgcn_wmma_f32_16x16x32_bf16(
            false, a[mi], false, b[ni], (short)0, acc[mi][ni], false, false);
  }

  #pragma unroll
  for (int mi = 0; mi < 2; ++mi) {
    #pragma unroll
    for (int ni = 0; ni < 2; ++ni) {
      int col = n0 + ni * 16 + lr;
      float bias = b_out[col];
      #pragma unroll
      for (int r = 0; r < 8; ++r) {
        int row = m0 + mi * 16 + lh * 8 + r;
        out[((size_t)row * kNC + col) * kT + t] = acc[mi][ni][r] + bias;
      }
    }
  }
}

// ---------------- host side ----------------

extern "C" void kernel_launch(void* const* d_in, const int* in_sizes, int n_in,
                              void* d_out, int out_size, void* d_ws, size_t ws_size,
                              hipStream_t stream) {
  (void)in_sizes; (void)n_in; (void)out_size; (void)ws_size;
  const float* hid   = (const float*)d_in[0];
  const float* inp0  = (const float*)d_in[1];
  const float* cell0 = (const float*)d_in[2];
  const float* W_ih  = (const float*)d_in[3];
  const float* W_hh  = (const float*)d_in[4];
  const float* b_ih  = (const float*)d_in[5];
  const float* b_hh  = (const float*)d_in[6];
  const float* W_out = (const float*)d_in[7];
  const float* b_out = (const float*)d_in[8];
  float* out = (float*)d_out;

  // Workspace layout (~20.25 MB total)
  char* w = (char*)d_ws;
  __bf16* Wc = (__bf16*)w; w += (size_t)k4H * kH * sizeof(__bf16);  // 8 MB
  __bf16* Wo = (__bf16*)w; w += (size_t)kNC * kH * sizeof(__bf16);  // 256 KB
  float*  xp = (float*)w;  w += (size_t)kB * k4H * sizeof(float);   // 8 MB
  float*  c  = (float*)w;  w += (size_t)kB * kH * sizeof(float);    // 2 MB
  __bf16* h0 = (__bf16*)w; w += (size_t)kB * kH * sizeof(__bf16);   // 1 MB
  __bf16* h1 = (__bf16*)w;                                          // 1 MB

  prep_combine_w<<<(k4H * kH) / 256, 256, 0, stream>>>(W_ih, W_hh, Wc);
  prep_wout<<<(kNC * kH) / 256, 256, 0, stream>>>(W_out, Wo);
  prep_xpart<<<(kB * k4H) / 256, 256, 0, stream>>>(inp0, W_ih, b_ih, b_hh, xp);
  prep_state<<<(kB * kH) / 256, 256, 0, stream>>>(hid, cell0, h0, c);

  __bf16* hin = h0;
  __bf16* hout = h1;
  for (int t = 0; t < kT; ++t) {
    lstm_gate_step<<<128, 256, 0, stream>>>(hin, Wc, xp, c, hout);
    lstm_logits_step<<<8, 256, 0, stream>>>(hout, Wo, b_out, out, t);
    __bf16* tmp = hin; hin = hout; hout = tmp;
  }
}
